// DecSwitchedDeconv_58720792871209
// MI455X (gfx1250) — matrix operand order repaired and checked by reading, not compiled, not run on hardware
//
#include <hip/hip_runtime.h>
#include <hip/hip_bf16.h>

typedef __attribute__((ext_vector_type(16))) _Float16 v16h;
typedef __attribute__((ext_vector_type(8)))  _Float16 v8h;
typedef __attribute__((ext_vector_type(8)))  float    v8f;

// Problem constants (match reference)
#define NSAMP 256
#define CIN   64
#define CMID  32
#define HW    32
#define HP    34          // padded spatial extent (1 halo each side)
#define BAND  4           // rows per band
#define NBAND (HW/BAND)   // 8 bands

#define W1H_ELEMS (9 * CMID * CIN)             // per-branch f16 weight slab (=18432)
#define W2H_ELEMS (9 * CIN * CMID)             // same count
#define HPAD_ELEMS ((size_t)NSAMP * HP * HP * CMID)
#define HPAD_BYTES (HPAD_ELEMS * 2)            // 18,939,904 B
#define WH_BYTES   ((size_t)8 * W1H_ELEMS * 2) // 294,912 B per weight set

// ---------------- CDNA5 async global<->LDS helpers (ASYNCcnt path) ----------------
// Inline asm: builtin pointer params are LangAS-qualified (cuda_device) and not
// spellable from HIP source, so emit the VGLOBAL async ops directly.
static __device__ __forceinline__ void async_g2l_b128(const void* g, void* l) {
    unsigned laddr = (unsigned)(unsigned long long)l;   // low 32 bits = LDS offset
    asm volatile("global_load_async_to_lds_b128 %0, %1, off"
                 :: "v"(laddr), "v"(g) : "memory");
}

static __device__ __forceinline__ void async_l2g_b128(void* g, const void* l) {
    unsigned laddr = (unsigned)(unsigned long long)l;
    asm volatile("global_store_async_from_lds_b128 %0, %1, off"
                 :: "v"(g), "v"(laddr) : "memory");
}

static __device__ __forceinline__ void async_wait0() {
#if __has_builtin(__builtin_amdgcn_s_wait_asynccnt)
    __builtin_amdgcn_s_wait_asynccnt(0);
#else
    asm volatile("s_wait_asynccnt 0" ::: "memory");
#endif
}

static __device__ __forceinline__ v16h ld16(const _Float16* p0, const _Float16* p1) {
    v8h lo = *(const v8h*)p0;
    v8h hi = *(const v8h*)p1;
    return __builtin_shufflevector(lo, hi, 0,1,2,3,4,5,6,7,8,9,10,11,12,13,14,15);
}

// -------------------------------------------------------------------------
// Prep: all 8 branches of W1/W2 -> flipped/transposed f16 GEMM layout in ws.
// w1h[nb][tap][o(32)][i(64)] = W1[nb][i][o][2-kh][2-kw]
// w2h[nb][tap][o(64)][i(32)] = W2[nb][i][o][2-kh][2-kw]
// -------------------------------------------------------------------------
__global__ __launch_bounds__(256) void prep_weights(
    const float* __restrict__ W1, const float* __restrict__ W2,
    _Float16* __restrict__ w1h, _Float16* __restrict__ w2h)
{
    const int nb = blockIdx.x;
    const int t  = threadIdx.x;
    for (int e = t; e < W1H_ELEMS; e += 256) {
        const int tap = e / (CMID * CIN);
        const int rem = e % (CMID * CIN);
        const int kh = tap / 3, kw = tap % 3;
        const int o1 = rem / CIN,  i1 = rem % CIN;   // layer1: o in 0..31, i in 0..63
        const int o2 = rem / CMID, i2 = rem % CMID;  // layer2: o in 0..63, i in 0..31
        w1h[nb * W1H_ELEMS + e] =
            (_Float16)W1[(((nb * CIN + i1) * CMID + o1) * 3 + (2 - kh)) * 3 + (2 - kw)];
        w2h[nb * W2H_ELEMS + e] =
            (_Float16)W2[(((nb * CMID + i2) * CIN + o2) * 3 + (2 - kh)) * 3 + (2 - kw)];
    }
}

// Zero the padded h tensor (halo stays zero; interior overwritten by conv1).
__global__ __launch_bounds__(256) void zero_hpad(uint4* __restrict__ p, unsigned n4)
{
    const unsigned stride = gridDim.x * 256u;
    uint4 zv = {0u, 0u, 0u, 0u};
    for (unsigned i = blockIdx.x * 256u + threadIdx.x; i < n4; i += stride) p[i] = zv;
}

// -------------------------------------------------------------------------
// Kernel 1: h = relu( conv3x3( relu(x), flipT(W1[idx]) ) + b1 ) -> padded f16 ws
// GEMM per tap: A = weights (M=16 out-ch x K=32 in-ch), B = pixels (K x N=16 px)
// -------------------------------------------------------------------------
__global__ __launch_bounds__(256) void conv1_wmma(
    const float* __restrict__ x, const int* __restrict__ yidx,
    const _Float16* __restrict__ w1h, const float* __restrict__ b1,
    _Float16* __restrict__ hpad)
{
    // [row 0..5][col 0..33][ch 0..63]; later reused as h tile [row4][col32][ch32]
    __shared__ __align__(16) _Float16 s_in[(BAND + 2) * HP * CIN];   // 26112 B
    __shared__ __align__(16) _Float16 s_w[W1H_ELEMS];                // 36864 B
    __shared__ float s_b[CMID];

    const int b    = blockIdx.x;
    const int band = blockIdx.y;
    const int r0   = band * BAND;
    const int t    = threadIdx.x;
    const int idx  = yidx[b];

    // ---- async-stage weights (contiguous f16 slab -> LDS, no VGPR round trip) ----
    {
        const _Float16* wsrc = w1h + idx * W1H_ELEMS;
        for (int e = t; e < W1H_ELEMS / 8; e += 256)          // 2304 x 16B
            async_g2l_b128(wsrc + e * 8, &s_w[e * 8]);
    }
    // ---- stage input band (coalesced global reads, relu, f16 convert) ----
    for (int e = t; e < (BAND + 2) * HP * CIN; e += 256) {
        const int ch = e / ((BAND + 2) * HP);
        const int rc = e % ((BAND + 2) * HP);
        const int rr = rc / HP, cc = rc % HP;
        const int gr = r0 + rr - 1, gc = cc - 1;
        float v = 0.f;
        if (gr >= 0 && gr < HW && gc >= 0 && gc < HW)
            v = x[((b * CIN + ch) * HW + gr) * HW + gc];
        v = v > 0.f ? v : 0.f;
        s_in[(rr * HP + cc) * CIN + ch] = (_Float16)v;
    }
    if (t < CMID) s_b[t] = b1[idx * CMID + t];
    async_wait0();
    __syncthreads();

    // ---- WMMA compute: wave -> one 16-pixel tile, both 16-ch M tiles ----
    const int wave = t >> 5, lane = t & 31;
    const int lh = lane >> 4;          // lane half (K-split select)
    const int ln = lane & 15;
    const int rw = wave >> 1;          // row within band (0..3)
    const int cb = (wave & 1) * 16;    // column base (0 or 16)
    const int col = cb + ln;

    v8f acc0 = {}, acc1 = {};
    for (int tap = 0; tap < 9; ++tap) {
        const int kh = tap / 3, kw = tap % 3;
        const _Float16* pin = &s_in[((rw + kh) * HP + (col + kw)) * CIN];
        #pragma unroll
        for (int kc = 0; kc < 2; ++kc) {            // K chunks of 32 in-ch
            const int kbase = kc * 32;
            v16h bf = *(const v16h*)(pin + kbase + lh * 16);
            const _Float16* pw0 = &s_w[(tap * CMID +  0 + ln) * CIN + kbase];
            const _Float16* pw1 = &s_w[(tap * CMID + 16 + ln) * CIN + kbase];
            v16h a0 = ld16(pw0 + lh * 16, pw0 + lh * 16 + 8);
            v16h a1 = ld16(pw1 + lh * 16, pw1 + lh * 16 + 8);
            acc0 = __builtin_amdgcn_wmma_f32_16x16x32_f16(false, a0, false, bf,
                                                          (short)0, acc0, false, false);
            acc1 = __builtin_amdgcn_wmma_f32_16x16x32_f16(false, a1, false, bf,
                                                          (short)0, acc1, false, false);
        }
    }

    // ---- epilogue: bias+relu into LDS tile, then async LDS->global rows ----
    __syncthreads();                    // everyone done reading s_in
    _Float16* s_h = s_in;               // reuse: [row4][col32][ch32] (8192 B)
    #pragma unroll
    for (int v = 0; v < 8; ++v) {
        const int c0 = v + lh * 8;      // D layout: M = v (+8 for upper lane half)
        float h0 = acc0[v] + s_b[c0];
        float h1 = acc1[v] + s_b[c0 + 16];
        h0 = h0 > 0.f ? h0 : 0.f;
        h1 = h1 > 0.f ? h1 : 0.f;
        s_h[(rw * HW + col) * CMID + c0]      = (_Float16)h0;
        s_h[(rw * HW + col) * CMID + c0 + 16] = (_Float16)h1;
    }
    __syncthreads();
    // interior rows r0+1..r0+4, cols 1..32: each row = 2048 B contiguous in hpad
    for (int e = t; e < 512; e += 256) {            // 512 x 16B
        const int rr  = e >> 7;                     // 0..3
        const int off = e & 127;
        _Float16* dst = hpad + ((size_t)(b * HP + r0 + 1 + rr) * HP + 1) * CMID + off * 8;
        async_l2g_b128(dst, s_h + rr * HW * CMID + off * 8);
    }
    // S_ENDPGM waits all counters; async stores complete before kernel2 runs.
}

// -------------------------------------------------------------------------
// Kernel 2: out = x + z * ( conv3x3( h, flipT(W2[idx]) ) + b2 )
// Input band + weights both arrive via async global->LDS copies.
// -------------------------------------------------------------------------
__global__ __launch_bounds__(256) void conv2_wmma(
    const float* __restrict__ x, const int* __restrict__ yidx,
    const float* __restrict__ z,
    const _Float16* __restrict__ w2h, const float* __restrict__ b2,
    const _Float16* __restrict__ hpad, float* __restrict__ out)
{
    __shared__ __align__(16) _Float16 s_in[(BAND + 2) * HP * CMID];  // 13056 B
    __shared__ __align__(16) _Float16 s_w[W2H_ELEMS];                // 36864 B
    __shared__ float s_b[CIN];
    __shared__ float s_z[CIN];

    const int b    = blockIdx.x;
    const int band = blockIdx.y;
    const int r0   = band * BAND;
    const int t    = threadIdx.x;
    const int idx  = yidx[b];

    // ---- async-stage padded-h band: ONE contiguous region, already [row][col][ch] ----
    {
        const _Float16* hsrc = hpad + (size_t)(b * HP + r0) * HP * CMID; // rows r0..r0+5
        for (int e = t; e < ((BAND + 2) * HP * CMID) / 8; e += 256)      // 816 x 16B
            async_g2l_b128(hsrc + e * 8, &s_in[e * 8]);
        const _Float16* wsrc = w2h + idx * W2H_ELEMS;
        for (int e = t; e < W2H_ELEMS / 8; e += 256)                     // 2304 x 16B
            async_g2l_b128(wsrc + e * 8, &s_w[e * 8]);
    }
    if (t < CIN) { s_b[t] = b2[idx * CIN + t]; s_z[t] = z[b * CIN + t]; }
    async_wait0();
    __syncthreads();

    const int wave = t >> 5, lane = t & 31;
    const int lh = lane >> 4;
    const int ln = lane & 15;
    const int rw = wave >> 1;
    const int cb = (wave & 1) * 16;
    const int col = cb + ln;

    v8f acc[4] = {{}, {}, {}, {}};   // 4 M-tiles of 16 output channels
    for (int tap = 0; tap < 9; ++tap) {
        const int kh = tap / 3, kw = tap % 3;
        const _Float16* pin = &s_in[((rw + kh) * HP + (col + kw)) * CMID];
        v16h bf = *(const v16h*)(pin + lh * 16);   // K = 32 (single chunk)
        #pragma unroll
        for (int mt = 0; mt < 4; ++mt) {
            const _Float16* pw = &s_w[(tap * CIN + mt * 16 + ln) * CMID];
            v16h a = ld16(pw + lh * 16, pw + lh * 16 + 8);
            acc[mt] = __builtin_amdgcn_wmma_f32_16x16x32_f16(false, a, false, bf,
                                                             (short)0, acc[mt], false, false);
        }
    }

    // ---- epilogue: out = x + z*(acc + b2), 64B-coalesced f32 stores ----
    const int row = r0 + rw;
    #pragma unroll
    for (int mt = 0; mt < 4; ++mt) {
        #pragma unroll
        for (int v = 0; v < 8; ++v) {
            const int ch = mt * 16 + v + lh * 8;
            const float val = acc[mt][v] + s_b[ch];
            const size_t o = ((size_t)(b * CIN + ch) * HW + row) * HW + col;
            out[o] = x[o] + s_z[ch] * val;
        }
    }
}

extern "C" void kernel_launch(void* const* d_in, const int* in_sizes, int n_in,
                              void* d_out, int out_size, void* d_ws, size_t ws_size,
                              hipStream_t stream) {
    const float* x    = (const float*)d_in[0];
    const int*   yidx = (const int*)d_in[1];   // y_index (harness: integer -> int)
    // d_in[2] = y_hard (unused by reference output)
    const float* z    = (const float*)d_in[3];
    const float* W1   = (const float*)d_in[4];
    const float* b1   = (const float*)d_in[5];
    const float* W2   = (const float*)d_in[6];
    const float* b2   = (const float*)d_in[7];
    float* out = (float*)d_out;

    char* ws = (char*)d_ws;
    _Float16* hpad = (_Float16*)ws;                              // 18,939,904 B
    _Float16* w1h  = (_Float16*)(ws + HPAD_BYTES);               //    294,912 B
    _Float16* w2h  = (_Float16*)(ws + HPAD_BYTES + WH_BYTES);    //    294,912 B

    zero_hpad<<<2048, 256, 0, stream>>>((uint4*)hpad, (unsigned)(HPAD_BYTES / 16));
    prep_weights<<<8, 256, 0, stream>>>(W1, W2, w1h, w2h);

    dim3 grid(NSAMP, NBAND);
    conv1_wmma<<<grid, 256, 0, stream>>>(x, yidx, w1h, b1, hpad);
    conv2_wmma<<<grid, 256, 0, stream>>>(x, yidx, z, w2h, b2, hpad, out);
    (void)in_sizes; (void)n_in; (void)out_size; (void)ws_size;
}
